// GAT_14508399526533
// MI455X (gfx1250) — compile-verified, matrix operand order checked by reading, not run on hardware
//
#include <hip/hip_runtime.h>
#include <hip/hip_bf16.h>

#define NNODES 50000
#define NEDGES 800000
#define EETOT  (NEDGES + NNODES)   // edges + self loops
#define INF_   256
#define HIDF   256
#define OUTF   128
#define NHEADS 4
#define CH     32
#define NIDX   4096
#define BN_EPS 1e-5f

typedef __attribute__((ext_vector_type(8)))  _Float16 v8h;
typedef __attribute__((ext_vector_type(16))) _Float16 v16h;
typedef __attribute__((ext_vector_type(8)))  float    v8f;

// ---------------------------------------------------------------------------
// Fragment loaders for V_WMMA_F32_16X16X32_F16 (wave32).
// 16-bit A 16x32 (MxK): lanes 0-15 hold row m0+lane, K = [k0,k0+8) in v0-3 and
// [k0+16,k0+24) in v4-7; lanes 16-31 hold K = [k0+8,k0+16) and [k0+24,k0+32).
// B 32x16 (KxN) mirrors this with N on lanes, so with W stored transposed
// (row n gives K contiguous) the same loader serves both operands.
// ---------------------------------------------------------------------------
__device__ inline v16h load_frag(const _Float16* __restrict__ base, int ld,
                                 int rc0, int k0, int lane) {
  int half = lane >> 4;
  int r    = lane & 15;
  const _Float16* p = base + (size_t)(rc0 + r) * ld + k0 + half * 8;
  v8h lo = *(const v8h*)(p);
  v8h hi = *(const v8h*)(p + 16);
  return __builtin_shufflevector(lo, hi, 0, 1, 2, 3, 4, 5, 6, 7,
                                 8, 9, 10, 11, 12, 13, 14, 15);
}

// ---------------------------------------------------------------------------
// WMMA GEMM: C[nRows x F_OUT] = A[nRows x F_IN] * Bt^T, f16 in / f32 out.
// Block = 128 threads = 4 waves. Each wave owns a 16x64 tile (4 accumulators).
// COL_GROUPS = F_OUT/64; waves cover (4/COL_GROUPS) row tiles per block.
// PROJ variant adds bias and accumulates per-column sum / sum-of-squares for
// the fused BatchNorm statistics.
// ---------------------------------------------------------------------------
template <int F_IN, int COL_GROUPS, bool PROJ>
__global__ void gemm_wmma(const _Float16* __restrict__ A,
                          const _Float16* __restrict__ Bt,
                          float* __restrict__ C, int nRowTiles,
                          const float* __restrict__ bias,
                          float* __restrict__ colsum,
                          float* __restrict__ colsumsq) {
  constexpr int F_OUT = COL_GROUPS * 64;
  const int wave = threadIdx.x >> 5;
  const int lane = threadIdx.x & 31;
  const int rowTile = blockIdx.x * (4 / COL_GROUPS) + (wave / COL_GROUPS);
  if (rowTile >= nRowTiles) return;  // wave-uniform: EXEC stays all-ones
  const int m0 = rowTile * 16;
  const int n0 = (wave % COL_GROUPS) * 64;

  v8f acc[4] = {};
#pragma unroll
  for (int k0 = 0; k0 < F_IN; k0 += 32) {
    v16h a = load_frag(A, F_IN, m0, k0, lane);
#pragma unroll
    for (int t = 0; t < 4; ++t) {
      v16h b = load_frag(Bt, F_IN, n0 + 16 * t, k0, lane);
      acc[t] = __builtin_amdgcn_wmma_f32_16x16x32_f16(
          false, a, false, b, (short)0, acc[t], false, false);
    }
  }

  // C/D layout: lane l, vgpr v -> row m0 + 8*(l>=16) + v, col n0+16t+(l&15)
  const int half = lane >> 4;
  const int r    = lane & 15;
#pragma unroll
  for (int t = 0; t < 4; ++t) {
    const int col = n0 + 16 * t + r;
    if (PROJ) {
      const float bv = bias[col];
      float s = 0.f, ss = 0.f;
#pragma unroll
      for (int v = 0; v < 8; ++v) {
        float val = acc[t][v] + bv;
        C[(size_t)(m0 + half * 8 + v) * F_OUT + col] = val;
        s += val;
        ss += val * val;
      }
      atomicAdd(&colsum[col], s);
      atomicAdd(&colsumsq[col], ss);
    } else {
#pragma unroll
      for (int v = 0; v < 8; ++v)
        C[(size_t)(m0 + half * 8 + v) * F_OUT + col] = acc[t][v];
    }
  }
}

// ---------------------------------------------------------------------------
// Small elementwise / reduction kernels
// ---------------------------------------------------------------------------
__global__ void cvt_f32_to_f16(const float* __restrict__ in,
                               _Float16* __restrict__ out, int n) {
  int t = blockIdx.x * blockDim.x + threadIdx.x;
  if (t < n) out[t] = (_Float16)in[t];
}

__global__ void cvt_transpose_f16(const float* __restrict__ W,
                                  _Float16* __restrict__ Wt, int rows, int cols) {
  int t = blockIdx.x * blockDim.x + threadIdx.x;
  if (t >= rows * cols) return;
  int r = t / cols, c = t % cols;
  Wt[(size_t)c * rows + r] = (_Float16)W[t];
}

__global__ void bn_stats(const float* __restrict__ colsum,
                         const float* __restrict__ colsumsq,
                         const float* __restrict__ gamma,
                         const float* __restrict__ beta,
                         float* __restrict__ scale, float* __restrict__ shift) {
  int c = threadIdx.x;  // 256 threads
  float mean = colsum[c] * (1.0f / NNODES);
  float var  = colsumsq[c] * (1.0f / NNODES) - mean * mean;
  float rstd = rsqrtf(var + BN_EPS);
  float sc   = rstd * gamma[c];
  scale[c] = sc;
  shift[c] = beta[c] - mean * sc;
}

__global__ void bn_apply(const float* __restrict__ h0,
                         const float* __restrict__ scale,
                         const float* __restrict__ shift,
                         _Float16* __restrict__ out) {
  int t = blockIdx.x * blockDim.x + threadIdx.x;
  if (t >= NNODES * HIDF) return;
  int c = t & (HIDF - 1);
  out[t] = (_Float16)(h0[t] * scale[c] + shift[c]);
}

__global__ void compute_alpha(const float* __restrict__ hW,
                              const float* __restrict__ a_src,
                              const float* __restrict__ a_dst,
                              float* __restrict__ as, float* __restrict__ ad) {
  int t = blockIdx.x * blockDim.x + threadIdx.x;  // n*NHEADS + h
  if (t >= NNODES * NHEADS) return;
  int h = t & (NHEADS - 1);
  const float* row = hW + (size_t)(t >> 2) * OUTF + h * CH;
  float s = 0.f, d = 0.f;
#pragma unroll
  for (int c = 0; c < CH; ++c) {
    float v = row[c];
    s += v * a_src[h * CH + c];
    d += v * a_dst[h * CH + c];
  }
  as[t] = s;
  ad[t] = d;
}

__global__ void out_init_bias(float* __restrict__ out,
                              const float* __restrict__ bias) {
  int t = blockIdx.x * blockDim.x + threadIdx.x;
  if (t < NNODES * OUTF) out[t] = bias[t & (OUTF - 1)];
}

// Order-preserving float -> uint mapping so atomicMax(uint) implements
// f32 segment-max (all inputs finite; init key 0 < any finite key).
__device__ inline unsigned mkey(float f) {
  unsigned u = __float_as_uint(f);
  return (u & 0x80000000u) ? ~u : (u | 0x80000000u);
}
__device__ inline float unmkey(unsigned k) {
  unsigned u = (k & 0x80000000u) ? (k & 0x7fffffffu) : ~k;
  return __uint_as_float(u);
}

__device__ inline void edge_sd(int e, const long long* src, const long long* dst,
                               int& s, int& d) {
  if (e < NEDGES) {
    s = (int)src[e];
    d = (int)dst[e];
  } else {  // appended self loop
    s = d = e - NEDGES;
  }
}

__global__ void edge_max(const long long* __restrict__ src,
                         const long long* __restrict__ dst,
                         const float* __restrict__ as,
                         const float* __restrict__ ad,
                         unsigned* __restrict__ maxkey) {
  int e = blockIdx.x * blockDim.x + threadIdx.x;
  if (e >= EETOT) return;
  int s, d;
  edge_sd(e, src, dst, s, d);
#pragma unroll
  for (int h = 0; h < NHEADS; ++h) {
    float v = as[s * NHEADS + h] + ad[d * NHEADS + h];
    v = (v > 0.f) ? v : 0.2f * v;  // leaky relu
    atomicMax(&maxkey[d * NHEADS + h], mkey(v));
  }
}

__global__ void edge_expsum(const long long* __restrict__ src,
                            const long long* __restrict__ dst,
                            const float* __restrict__ as,
                            const float* __restrict__ ad,
                            const unsigned* __restrict__ maxkey,
                            float* __restrict__ wedge,
                            float* __restrict__ denom) {
  int e = blockIdx.x * blockDim.x + threadIdx.x;
  if (e >= EETOT) return;
  int s, d;
  edge_sd(e, src, dst, s, d);
#pragma unroll
  for (int h = 0; h < NHEADS; ++h) {
    float v = as[s * NHEADS + h] + ad[d * NHEADS + h];
    v = (v > 0.f) ? v : 0.2f * v;
    float w = __expf(v - unmkey(maxkey[d * NHEADS + h]));
    wedge[(size_t)e * NHEADS + h] = w;
    atomicAdd(&denom[d * NHEADS + h], w);
  }
}

// One thread per (edge, head): gathers a 32-float chunk of hW[src]
// (L2-resident: 25.6 MB << 192 MB L2) and scatters via f32 atomics.
__global__ void edge_aggr(const long long* __restrict__ src,
                          const long long* __restrict__ dst,
                          const float* __restrict__ wedge,
                          const float* __restrict__ denom,
                          const float* __restrict__ hW,
                          float* __restrict__ out) {
  int t = blockIdx.x * blockDim.x + threadIdx.x;
  if (t >= EETOT * NHEADS) return;
  int e = t >> 2, h = t & (NHEADS - 1);
  int s, d;
  edge_sd(e, src, dst, s, d);
  float alpha = wedge[t] / denom[d * NHEADS + h];
  const float* hr = hW + (size_t)s * OUTF + h * CH;
  float* orow = out + (size_t)d * OUTF + h * CH;
#pragma unroll
  for (int c = 0; c < CH; ++c) atomicAdd(&orow[c], hr[c] * alpha);
}

__global__ void gather_rows(const float* __restrict__ h,
                            const long long* __restrict__ idx,
                            float* __restrict__ out) {
  int t = blockIdx.x * blockDim.x + threadIdx.x;
  if (t >= NIDX * OUTF) return;
  int i = t >> 7;
  int j = t & (OUTF - 1);
  out[t] = h[(size_t)idx[i] * OUTF + j];
}

// ---------------------------------------------------------------------------
// Host-side orchestration
// ---------------------------------------------------------------------------
static inline int cdiv(int a, int b) { return (a + b - 1) / b; }

extern "C" void kernel_launch(void* const* d_in, const int* in_sizes, int n_in,
                              void* d_out, int out_size, void* d_ws, size_t ws_size,
                              hipStream_t stream) {
  (void)in_sizes; (void)n_in; (void)out_size; (void)ws_size;
  // inputs (setup_inputs order)
  const float*     x      = (const float*)d_in[0];
  const long long* ei     = (const long long*)d_in[1];   // [2, E] int64
  const long long* idx    = (const long long*)d_in[2];   // [4096] int64
  const float*     proj_w = (const float*)d_in[3];
  const float*     proj_b = (const float*)d_in[4];
  const float*     gamma  = (const float*)d_in[5];
  const float*     beta   = (const float*)d_in[6];
  const float*     w0     = (const float*)d_in[7];
  const float*     a_src0 = (const float*)d_in[8];
  const float*     a_dst0 = (const float*)d_in[9];
  const float*     b0     = (const float*)d_in[10];
  const float*     w1     = (const float*)d_in[11];
  const float*     a_src1 = (const float*)d_in[12];
  const float*     a_dst1 = (const float*)d_in[13];
  const float*     b1     = (const float*)d_in[14];
  const long long* srcE   = ei;
  const long long* dstE   = ei + NEDGES;
  float* outp = (float*)d_out;

  // workspace carve-up (256B aligned)
  char* base = (char*)d_ws;
  size_t off = 0;
  auto take = [&](size_t bytes) -> void* {
    void* p = base + off;
    off = (off + bytes + 255) & ~(size_t)255;
    return p;
  };
  float*    h0      = (float*)take((size_t)NNODES * HIDF * 4);  // 51.2MB; reused as hW|out
  _Float16* xh      = (_Float16*)take((size_t)NNODES * HIDF * 2); // reused as hbn, h1f16
  float*    wedge   = (float*)take((size_t)EETOT * NHEADS * 4);
  float*    asv     = (float*)take((size_t)NNODES * NHEADS * 4);
  float*    adv     = (float*)take((size_t)NNODES * NHEADS * 4);
  unsigned* maxkey  = (unsigned*)take((size_t)NNODES * NHEADS * 4);
  float*    denom   = (float*)take((size_t)NNODES * NHEADS * 4);
  _Float16* pwT     = (_Float16*)take((size_t)HIDF * INF_ * 2);
  _Float16* w0T     = (_Float16*)take((size_t)OUTF * HIDF * 2);
  _Float16* w1T     = (_Float16*)take((size_t)OUTF * OUTF * 2);
  float*    colsum  = (float*)take(HIDF * 4);
  float*    colsqs  = (float*)take(HIDF * 4);
  float*    bscale  = (float*)take(HIDF * 4);
  float*    bshift  = (float*)take(HIDF * 4);
  // aliases into the h0 region once BN has consumed it:
  float* hW   = h0;                         // [N, 128]
  float* aggr = h0 + (size_t)NNODES * OUTF; // [N, 128]

  const int TB = 256;
  const int rowTiles = NNODES / 16;  // 3125

  // --- prep: f16 weights (transposed) + f16 input ---
  hipMemsetAsync(colsum, 0, HIDF * 4, stream);
  hipMemsetAsync(colsqs, 0, HIDF * 4, stream);
  cvt_transpose_f16<<<cdiv(INF_ * HIDF, TB), TB, 0, stream>>>(proj_w, pwT, INF_, HIDF);
  cvt_transpose_f16<<<cdiv(HIDF * OUTF, TB), TB, 0, stream>>>(w0, w0T, HIDF, OUTF);
  cvt_transpose_f16<<<cdiv(OUTF * OUTF, TB), TB, 0, stream>>>(w1, w1T, OUTF, OUTF);
  cvt_f32_to_f16<<<cdiv(NNODES * INF_, TB), TB, 0, stream>>>(x, xh, NNODES * INF_);

  // --- proj GEMM (WMMA) + fused BN statistics ---
  gemm_wmma<INF_, 4, true><<<rowTiles, 128, 0, stream>>>(
      xh, pwT, h0, rowTiles, proj_b, colsum, colsqs);
  bn_stats<<<1, HIDF, 0, stream>>>(colsum, colsqs, gamma, beta, bscale, bshift);
  bn_apply<<<cdiv(NNODES * HIDF, TB), TB, 0, stream>>>(h0, bscale, bshift, xh);

  // --- GAT layer 0 ---
  gemm_wmma<HIDF, 2, false><<<cdiv(rowTiles, 2), 128, 0, stream>>>(
      xh, w0T, hW, rowTiles, nullptr, nullptr, nullptr);
  compute_alpha<<<cdiv(NNODES * NHEADS, TB), TB, 0, stream>>>(hW, a_src0, a_dst0, asv, adv);
  hipMemsetAsync(maxkey, 0, (size_t)NNODES * NHEADS * 4, stream);
  hipMemsetAsync(denom, 0, (size_t)NNODES * NHEADS * 4, stream);
  out_init_bias<<<cdiv(NNODES * OUTF, TB), TB, 0, stream>>>(aggr, b0);
  edge_max<<<cdiv(EETOT, TB), TB, 0, stream>>>(srcE, dstE, asv, adv, maxkey);
  edge_expsum<<<cdiv(EETOT, TB), TB, 0, stream>>>(srcE, dstE, asv, adv, maxkey, wedge, denom);
  edge_aggr<<<cdiv(EETOT * NHEADS, TB), TB, 0, stream>>>(srcE, dstE, wedge, denom, hW, aggr);

  // --- GAT layer 1 (input = layer0 output, converted to f16) ---
  cvt_f32_to_f16<<<cdiv(NNODES * OUTF, TB), TB, 0, stream>>>(aggr, xh, NNODES * OUTF);
  gemm_wmma<OUTF, 2, false><<<cdiv(rowTiles, 2), 128, 0, stream>>>(
      xh, w1T, hW, rowTiles, nullptr, nullptr, nullptr);
  compute_alpha<<<cdiv(NNODES * NHEADS, TB), TB, 0, stream>>>(hW, a_src1, a_dst1, asv, adv);
  hipMemsetAsync(maxkey, 0, (size_t)NNODES * NHEADS * 4, stream);
  hipMemsetAsync(denom, 0, (size_t)NNODES * NHEADS * 4, stream);
  out_init_bias<<<cdiv(NNODES * OUTF, TB), TB, 0, stream>>>(aggr, b1);
  edge_max<<<cdiv(EETOT, TB), TB, 0, stream>>>(srcE, dstE, asv, adv, maxkey);
  edge_expsum<<<cdiv(EETOT, TB), TB, 0, stream>>>(srcE, dstE, asv, adv, maxkey, wedge, denom);
  edge_aggr<<<cdiv(EETOT * NHEADS, TB), TB, 0, stream>>>(srcE, dstE, wedge, denom, hW, aggr);

  // --- final gather ---
  gather_rows<<<cdiv(NIDX * OUTF, TB), TB, 0, stream>>>(aggr, idx, outp);
}
